// Attention_80324478369916
// MI455X (gfx1250) — matrix operand order repaired and checked by reading, not compiled, not run on hardware
//
#include <hip/hip_runtime.h>

// Problem (from reference, after the raw reshape):
//   xf[B=128][N=256][D=128] f32, w_qkv[1536][128], w_out[128][512], b_out[128]
//   HEADS=8, DIM_HEAD=64, top-k = int(256*0.7)=179, scale = 0.125
#define BATCH   128
#define NTOK    256
#define DMODEL  128
#define HEADS   8
#define DHEAD   64
#define INNER   512
#define KTOPK   179
#define SCALE   0.125f

typedef __attribute__((ext_vector_type(16))) _Float16 v16h;
typedef __attribute__((ext_vector_type(8)))  _Float16 v8h;
typedef __attribute__((ext_vector_type(8)))  float    v8f;
typedef __attribute__((ext_vector_type(4)))  int      v4i;

#if __has_builtin(__builtin_amdgcn_global_load_async_to_lds_b128)
#define HAVE_ASYNC_LDS 1
#else
#define HAVE_ASYNC_LDS 0
#endif

// 16B global -> LDS copy; async (ASYNCcnt-tracked) when the builtin exists.
// Builtin signature (from hipcc diagnostic): (AS1 int4*, AS3 int4*, imm, imm).
__device__ __forceinline__ void copy16_g2l(const _Float16* g, _Float16* l) {
#if HAVE_ASYNC_LDS
    __builtin_amdgcn_global_load_async_to_lds_b128(
        (__attribute__((address_space(1))) v4i*)g,
        (__attribute__((address_space(3))) v4i*)l, 0, 0);
#else
    *(v8h*)l = *(const v8h*)g;
#endif
}

__device__ __forceinline__ void wait_async_zero() {
#if HAVE_ASYNC_LDS
#if __has_builtin(__builtin_amdgcn_s_wait_asynccnt)
    __builtin_amdgcn_s_wait_asynccnt(0);
#else
    asm volatile("s_wait_asynccnt 0" ::: "memory");
#endif
#endif
}

__device__ __forceinline__ float wave_sum(float v) {
    #pragma unroll
    for (int m = 16; m >= 1; m >>= 1) v += __shfl_xor(v, m, 32);
    return v;
}
__device__ __forceinline__ float wave_max(float v) {
    #pragma unroll
    for (int m = 16; m >= 1; m >>= 1) v = fmaxf(v, __shfl_xor(v, m, 32));
    return v;
}
__device__ __forceinline__ int wave_sumi(int v) {
    #pragma unroll
    for (int m = 16; m >= 1; m >>= 1) v += __shfl_xor(v, m, 32);
    return v;
}

__device__ __forceinline__ v8f wmma_f16(v16h a, v16h b, v8f c) {
    return __builtin_amdgcn_wmma_f32_16x16x32_f16(false, a, false, b,
                                                  (short)0, c, false, false);
}

// A-fragment (16x32 f16) from a row-major f16 row pointer:
//   lane m in [0,16): K {0..7,16..23}; lanes 16-31: K {8..15,24..31}
__device__ __forceinline__ v16h load_a_frag(const _Float16* row, int ka) {
    const v8h a0 = *(const v8h*)(row + ka);
    const v8h a1 = *(const v8h*)(row + ka + 16);
    v16h a;
    #pragma unroll
    for (int i = 0; i < 8; ++i) { a[i] = a0[i]; a[8 + i] = a1[i]; }
    return a;
}
// B-fragment (32x16 f16): lane = column; lanes 0-15: K {0..7,16..23}; lanes 16-31: K {8..15,24..31}
__device__ __forceinline__ v16h load_b_frag(const _Float16* col, int kb) {
    const v8h b0 = *(const v8h*)(col + kb);
    const v8h b1 = *(const v8h*)(col + kb + 16);
    v16h b;
    #pragma unroll
    for (int i = 0; i < 8; ++i) { b[i] = b0[i]; b[8 + i] = b1[i]; }
    return b;
}

// ---------------------------------------------------------------------------
// Kernel 0: one-shot f32 -> f16 conversion (x, w_qkv, w_out).
// ---------------------------------------------------------------------------
__global__ void cvt_f16_kernel(const float* __restrict__ src,
                               _Float16* __restrict__ dst, int n)
{
    const int i = (blockIdx.x * blockDim.x + threadIdx.x) * 4;
    if (i + 3 < n) {
        const float4 v = *(const float4*)(src + i);
        dst[i]     = (_Float16)v.x;
        dst[i + 1] = (_Float16)v.y;
        dst[i + 2] = (_Float16)v.z;
        dst[i + 3] = (_Float16)v.w;
    } else {
        for (int j = i; j < n; ++j) dst[j] = (_Float16)src[j];
    }
}

// ---------------------------------------------------------------------------
// Kernel 1: QKV projection.  x16[B*N,128] @ wq16^T[128,1536] -> per-head f16
// Q/K/V [B][H][N][64].  One 16x16 tile per wave, K-loop 4 x 32, all-f16 loads.
// ---------------------------------------------------------------------------
__global__ void qkv_proj_kernel(const _Float16* __restrict__ x16,
                                const _Float16* __restrict__ wq16,
                                _Float16* __restrict__ Qh,
                                _Float16* __restrict__ Kh,
                                _Float16* __restrict__ Vh)
{
    const int wave = threadIdx.x >> 5;
    const int lane = threadIdx.x & 31;
    const int half = lane >> 4;
    const int l16  = lane & 15;

    long tile = (long)blockIdx.x * 8 + wave;       // B * 16 * 96 tiles
    const int et = (int)(tile % 96);
    long r = tile / 96;
    const int nt = (int)(r % 16);
    const int b  = (int)(r / 16);
    const int tok0 = nt * 16, e0 = et * 16;

    const _Float16* arow = x16  + ((long)b * NTOK + tok0 + l16) * DMODEL;
    const _Float16* brow = wq16 + (long)(e0 + l16) * DMODEL;

    v8f acc = {};
    #pragma unroll
    for (int ks = 0; ks < 4; ++ks) {
        const v16h a  = load_a_frag(arow, ks * 32 + half * 8);
        const v16h bf = load_b_frag(brow, ks * 32 + half * 8);
        acc = wmma_f16(a, bf, acc);
    }

    const int feat  = e0 + l16;
    const int which = feat >> 9;
    const int hd    = feat & 511;
    const int h     = hd >> 6;
    const int d     = hd & 63;
    _Float16* dst = (which == 0) ? Qh : (which == 1) ? Kh : Vh;
    #pragma unroll
    for (int i = 0; i < 8; ++i) {
        const int tok = tok0 + half * 8 + i;
        dst[(((long)b * HEADS + h) * NTOK + tok) * DHEAD + d] = (_Float16)acc[i];
    }
}

// ---------------------------------------------------------------------------
// Kernel 2: attention for one (b,h) x 64-row tile.
//   async-stage V into LDS while computing S = QK^T*scale (WMMA)
//   transpose V in LDS; radix-select-179 + masked softmax -> P (f16, LDS)
//   O = P V via WMMA with pure 16B LDS fragment loads.
//   LDS: S 64KB + Vr/P union 32KB + Vt 32KB = 128KB (<= 320KB/WGP).
// ---------------------------------------------------------------------------
__global__ void attn_kernel(const _Float16* __restrict__ Qh,
                            const _Float16* __restrict__ Kh,
                            const _Float16* __restrict__ Vh,
                            _Float16* __restrict__ Oh)
{
    __shared__ float    S[64 * 256];        // scores (f32)
    __shared__ _Float16 VrP[NTOK * DHEAD];  // V row-major staging, then P f16
    __shared__ _Float16 Vt[DHEAD * NTOK];   // V transposed: Vt[d][k]

    const int wave = threadIdx.x >> 5;
    const int lane = threadIdx.x & 31;
    const int half = lane >> 4;
    const int l16  = lane & 15;

    const int blk = blockIdx.x;             // (b*8+h)*4 + rowTile
    const int rt  = blk & 3;
    const int bh  = blk >> 2;
    const long base = (long)bh * NTOK * DHEAD;
    const int row0g = rt * 64;

    // --- stage V (32KB) into LDS; async when available ----------------------
    const _Float16* Vg = Vh + base;
    #pragma unroll
    for (int it = 0; it < 8; ++it) {
        const int c = threadIdx.x + it * 256;   // 2048 x 16B chunks
        copy16_g2l(Vg + c * 8, &VrP[c * 8]);
    }

    // --- S = Q K^T * scale : 64 tiles of 16x16, 8 per wave ------------------
    #pragma unroll
    for (int t = 0; t < 8; ++t) {
        const int tile = wave * 8 + t;
        const int tr = tile >> 4;               // 0..3
        const int tc = tile & 15;               // 0..15
        const _Float16* qrow = Qh + base + (long)(row0g + tr * 16 + l16) * DHEAD;
        const _Float16* krow = Kh + base + (long)(tc * 16 + l16) * DHEAD;
        v8f acc = {};
        #pragma unroll
        for (int ks = 0; ks < 2; ++ks) {
            const v16h a  = load_a_frag(qrow, ks * 32 + half * 8);
            const v16h bf = load_b_frag(krow, ks * 32 + half * 8);
            acc = wmma_f16(a, bf, acc);
        }
        #pragma unroll
        for (int i = 0; i < 8; ++i)
            S[(tr * 16 + half * 8 + i) * 256 + tc * 16 + l16] = acc[i] * SCALE;
    }

    wait_async_zero();
    __syncthreads();            // S complete, Vr staged

    // --- transpose V: Vr[k][d] -> Vt[d][k] ----------------------------------
    #pragma unroll
    for (int it = 0; it < 8; ++it) {
        const int g  = threadIdx.x + it * 256;  // 2048 groups of 8 halfs
        const int k  = g >> 3;
        const int d0 = (g & 7) * 8;
        const v8h v = *(const v8h*)(&VrP[k * DHEAD + d0]);
        #pragma unroll
        for (int j = 0; j < 8; ++j) Vt[(d0 + j) * 256 + k] = v[j];
    }
    __syncthreads();            // Vt ready; Vr region free for P

    _Float16* P = VrP;          // reuse staging buffer as P[64][256] f16

    // --- per-row top-k threshold + masked softmax, 8 rows per wave ----------
    #pragma unroll 1
    for (int r = 0; r < 8; ++r) {
        const int row = wave * 8 + r;
        float v[8]; unsigned key[8];
        #pragma unroll
        for (int j = 0; j < 8; ++j) {
            v[j] = S[row * 256 + lane * 8 + j];
            const unsigned u = __float_as_uint(v[j]);
            key[j] = (u & 0x80000000u) ? ~u : (u | 0x80000000u);
        }
        unsigned prefix = 0; int kk = KTOPK;
        #pragma unroll 1
        for (int bit = 31; bit >= 0; --bit) {
            const unsigned cand  = prefix | (1u << bit);
            const unsigned maskh = ~((1u << bit) - 1u);
            int c = 0;
            #pragma unroll
            for (int j = 0; j < 8; ++j) c += ((key[j] & maskh) == cand);
            c = wave_sumi(c);
            if (kk <= c) prefix = cand; else kk -= c;
        }
        const float thr = __uint_as_float((prefix & 0x80000000u)
                                          ? (prefix ^ 0x80000000u) : ~prefix);
        float m = -1e30f;
        #pragma unroll
        for (int j = 0; j < 8; ++j) m = fmaxf(m, v[j]);
        m = wave_max(m);
        float e[8]; float s = 0.f;
        #pragma unroll
        for (int j = 0; j < 8; ++j) {
            e[j] = (v[j] >= thr) ? __expf(v[j] - m) : 0.f;
            s += e[j];
        }
        s = wave_sum(s);
        const float inv = 1.f / s;
        v8h pv;
        #pragma unroll
        for (int j = 0; j < 8; ++j) pv[j] = (_Float16)(e[j] * inv);
        *(v8h*)(&P[row * 256 + lane * 8]) = pv;   // one 16B LDS store
    }
    __syncthreads();

    // --- O = P V : 16 tiles of 16x16, 2 per wave, K = 256 -------------------
    const int bq = bh >> 3;
    const int h  = bh & 7;
    #pragma unroll
    for (int t = 0; t < 2; ++t) {
        const int tile = wave + t * 8;
        const int tr = tile >> 2;
        const int tc = tile & 3;
        const int arow = tr * 16 + l16;
        const int dcol = tc * 16 + l16;
        v8f acc = {};
        #pragma unroll
        for (int ks = 0; ks < 8; ++ks) {
            const v16h a  = load_a_frag(&P[arow * 256], ks * 32 + half * 8);
            const v16h bf = load_b_frag(&Vt[dcol * 256], ks * 32 + half * 8);
            acc = wmma_f16(a, bf, acc);
        }
        #pragma unroll
        for (int i = 0; i < 8; ++i) {
            const int tok = row0g + tr * 16 + half * 8 + i;
            Oh[((long)bq * NTOK + tok) * INNER + h * DHEAD + dcol] = (_Float16)acc[i];
        }
    }
}

// ---------------------------------------------------------------------------
// Kernel 3: output projection.  O[32768,512] @ wo16^T[512,128] + b_out.
// ---------------------------------------------------------------------------
__global__ void out_proj_kernel(const _Float16* __restrict__ Oh,
                                const _Float16* __restrict__ wo16,
                                const float* __restrict__ bout,
                                float* __restrict__ out)
{
    const int wave = threadIdx.x >> 5;
    const int lane = threadIdx.x & 31;
    const int half = lane >> 4;
    const int l16  = lane & 15;

    long tile = (long)blockIdx.x * 8 + wave;   // 2048 row x 8 col tiles
    const int ct = (int)(tile & 7);
    const long rt = tile >> 3;
    const int row0 = (int)(rt * 16);

    const _Float16* arow = Oh   + (long)(row0 + l16) * INNER;
    const _Float16* brow = wo16 + (long)(ct * 16 + l16) * INNER;

    v8f acc = {};
    #pragma unroll
    for (int ks = 0; ks < 16; ++ks) {
        const v16h a  = load_a_frag(arow, ks * 32 + half * 8);
        const v16h bf = load_b_frag(brow, ks * 32 + half * 8);
        acc = wmma_f16(a, bf, acc);
    }

    const int col = ct * 16 + l16;
    const float bb = bout[col];
    #pragma unroll
    for (int i = 0; i < 8; ++i) {
        const int row = row0 + half * 8 + i;
        out[(long)row * DMODEL + col] = acc[i] + bb;
    }
}

// ---------------------------------------------------------------------------
extern "C" void kernel_launch(void* const* d_in, const int* in_sizes, int n_in,
                              void* d_out, int out_size, void* d_ws, size_t ws_size,
                              hipStream_t stream)
{
    const float* x    = (const float*)d_in[0];
    const float* wqkv = (const float*)d_in[1];
    const float* wout = (const float*)d_in[2];
    const float* bout = (const float*)d_in[3];
    float* out = (float*)d_out;

    const long HEAD_ELEMS = (long)BATCH * HEADS * NTOK * DHEAD;   // 16,777,216
    const long X_ELEMS  = (long)BATCH * NTOK * DMODEL;            //  4,194,304
    const long WQ_ELEMS = 3L * INNER * DMODEL;                    //    196,608
    const long WO_ELEMS = (long)DMODEL * INNER;                   //     65,536

    _Float16* Qh  = (_Float16*)d_ws;
    _Float16* Kh  = Qh  + HEAD_ELEMS;
    _Float16* Vh  = Kh  + HEAD_ELEMS;
    _Float16* Oh  = Vh  + HEAD_ELEMS;
    _Float16* X16 = Oh  + HEAD_ELEMS;
    _Float16* Wq16 = X16 + X_ELEMS;
    _Float16* Wo16 = Wq16 + WQ_ELEMS;

    // 0) one-shot f32->f16 conversion (memory-bound, ~9MB total)
    cvt_f16_kernel<<<(int)(X_ELEMS  / 1024), 256, 0, stream>>>(x,    X16,  (int)X_ELEMS);
    cvt_f16_kernel<<<(int)(WQ_ELEMS / 1024), 256, 0, stream>>>(wqkv, Wq16, (int)WQ_ELEMS);
    cvt_f16_kernel<<<(int)(WO_ELEMS / 1024), 256, 0, stream>>>(wout, Wo16, (int)WO_ELEMS);

    // 1) QKV projection: 196608 tiles / 8 waves -> 24576 blocks
    qkv_proj_kernel<<<24576, 256, 0, stream>>>(X16, Wq16, Qh, Kh, Vh);

    // 2) Attention: (B*H)*4 row tiles = 4096 blocks (128KB LDS each)
    attn_kernel<<<4096, 256, 0, stream>>>(Qh, Kh, Vh, Oh);

    // 3) Output projection: 16384 tiles / 8 waves -> 2048 blocks
    out_proj_kernel<<<2048, 256, 0, stream>>>(Oh, Wo16, bout, out);
}
